// ManualAttention_79731772883737
// MI455X (gfx1250) — matrix operand order repaired and checked by reading, not compiled, not run on hardware
//
#include <hip/hip_runtime.h>
#include <hip/hip_bf16.h>

#define DIMN  1024
#define SEQ   2048
#define BATCH 4
#define NH    16
#define HD    64

typedef __attribute__((ext_vector_type(16))) __bf16 v16bf;
typedef __attribute__((ext_vector_type(8)))  __bf16 v8bf;
typedef __attribute__((ext_vector_type(8)))  float  v8f;
typedef unsigned int u32x4 __attribute__((ext_vector_type(4)));
typedef int          i32x4 __attribute__((ext_vector_type(4)));
typedef int          i32x8 __attribute__((ext_vector_type(8)));

static __device__ __forceinline__ __bf16 f2bf(float f) { return (__bf16)f; }

// A-matrix 16x32 bf16 fragment: lane holds row (lane&15),
// K = half*16 .. half*16+15 contiguous.
static __device__ __forceinline__ v16bf fragA_bf16(const __bf16* prow, int k0, int half) {
  return *(const v16bf*)(prow + k0 + half * 16);
}

// B-matrix 32x16 bf16 fragment: lane holds column (lane&15),
// K = half*16 .. half*16+15 contiguous.
static __device__ __forceinline__ v16bf fragB_bf16(const __bf16* prow, int k0, int half) {
  return *(const v16bf*)(prow + k0 + half * 16);
}

static __device__ __forceinline__ v8f wmma_bf16(v16bf a, v16bf b, v8f c) {
  return __builtin_amdgcn_wmma_f32_16x16x32_bf16(false, a, false, b,
                                                 (short)0, c, false, false);
}

// Async copy 16B global -> LDS (gfx1250, tracked by ASYNCcnt).
static __device__ __forceinline__ void async_cp16(__bf16* lds_dst, const __bf16* gsrc) {
  const unsigned loff = (unsigned)(size_t)lds_dst;  // low 32 bits = LDS byte addr
  asm volatile("global_load_async_to_lds_b128 %0, %1, off"
               :: "v"(loff), "v"(gsrc)
               : "memory");
}

static __device__ __forceinline__ void wait_async0() {
  asm volatile("s_wait_asynccnt 0" ::: "memory");
}

// TDM 2-D tile load global -> LDS (gfx1250 TENSOR_LOAD_TO_LDS, TENSORcnt).
// D# per CDNA5 ISA: 2-byte elements, 2-D tensor,
// tile = tile_d1 rows x tile_d0 elements, row stride d0_stride elements.
// 6-arg builtin form (clang-23): (g0, g1, g2, g3, <zero i32x8>, cpol).
static __device__ __forceinline__ void tdm_load_2d(
    unsigned lds_addr, const void* gaddr,
    unsigned tensor_d0, unsigned tensor_d1, unsigned d0_stride,
    unsigned tile_d0, unsigned tile_d1) {
  const unsigned long long ga = (unsigned long long)(size_t)gaddr;
  u32x4 g0;
  g0[0] = 1u;                                              // count=1 (valid)
  g0[1] = lds_addr;                                        // LDS byte address
  g0[2] = (unsigned)(ga & 0xFFFFFFFFu);                    // global_addr[31:0]
  g0[3] = (unsigned)((ga >> 32) & 0x1FFFFFFu) | (2u << 30);// addr[56:32] | type=2
  i32x8 g1;
  g1[0] = (int)(1u << 16);                                 // data_size=1 (2B)
  g1[1] = (int)((tensor_d0 & 0xFFFFu) << 16);              // tensor_dim0 lo
  g1[2] = (int)(((tensor_d0 >> 16) & 0xFFFFu) |
                ((tensor_d1 & 0xFFFFu) << 16));            // dim0 hi | dim1 lo
  g1[3] = (int)(((tensor_d1 >> 16) & 0xFFFFu) |
                ((tile_d0 & 0xFFFFu) << 16));              // dim1 hi | tile_d0
  g1[4] = (int)(tile_d1 & 0xFFFFu);                        // tile_d1, tile_d2=0
  g1[5] = (int)d0_stride;                                  // dim0_stride[31:0]
  g1[6] = 0;                                               // stride hi, dim1_stride lo
  g1[7] = 0;
  i32x4 g2 = {0, 0, 0, 0};
  i32x4 g3 = {0, 0, 0, 0};
  i32x8 g4 = {0, 0, 0, 0, 0, 0, 0, 0};
  __builtin_amdgcn_tensor_load_to_lds(g0, g1, g2, g3, g4, 0);
}

// ---------------------------------------------------------------------------
// Kernel 0: one-shot fp32 -> bf16 conversion (x, qkv_w, out_w), 8 elts/thread.
// ---------------------------------------------------------------------------
__global__ __launch_bounds__(256) void cvt_kernel(const float* __restrict__ src,
                                                  __bf16* __restrict__ dst) {
  const size_t i = ((size_t)blockIdx.x * blockDim.x + threadIdx.x) * 8;
  float4 a = *(const float4*)(src + i);
  float4 b = *(const float4*)(src + i + 4);
  v8bf o;
  o[0] = f2bf(a.x); o[1] = f2bf(a.y); o[2] = f2bf(a.z); o[3] = f2bf(a.w);
  o[4] = f2bf(b.x); o[5] = f2bf(b.y); o[6] = f2bf(b.z); o[7] = f2bf(b.w);
  *(v8bf*)(dst + i) = o;
}

// ---------------------------------------------------------------------------
// Kernel 1: qkv = x @ qkv_w^T + qkv_b (all-bf16 operands, f32 accum)
//           -> Q,K bf16 [B,H,S,64], V^T bf16 [B,H,64,S]
// Wave tile: 32(M) x 64(N); block = 8 waves covering 256(M) x 64(N).
// ---------------------------------------------------------------------------
__global__ __launch_bounds__(256) void qkv_kernel(
    const __bf16* __restrict__ x, const __bf16* __restrict__ w,
    const float* __restrict__ bias,
    __bf16* __restrict__ qb, __bf16* __restrict__ kb, __bf16* __restrict__ vtb) {
  const int lane = threadIdx.x & 31, wave = threadIdx.x >> 5;
  const int r16 = lane & 15, half = lane >> 4;
  const int m0 = blockIdx.y * 256 + wave * 32;
  const int n0 = blockIdx.x * 64;

  const __bf16* arow[2];
  const __bf16* brow[4];
#pragma unroll
  for (int mt = 0; mt < 2; ++mt)
    arow[mt] = x + (size_t)(m0 + mt * 16 + r16) * DIMN;
#pragma unroll
  for (int nt = 0; nt < 4; ++nt)
    brow[nt] = w + (size_t)(n0 + nt * 16 + r16) * DIMN;

  v8f acc[2][4] = {};
  for (int k0 = 0; k0 < DIMN; k0 += 32) {
    v16bf a[2], bf[4];
#pragma unroll
    for (int mt = 0; mt < 2; ++mt) {
      a[mt] = fragA_bf16(arow[mt], k0, half);
      __builtin_prefetch(arow[mt] + k0 + 64, 0, 1);
    }
#pragma unroll
    for (int nt = 0; nt < 4; ++nt) {
      bf[nt] = fragB_bf16(brow[nt], k0, half);
      __builtin_prefetch(brow[nt] + k0 + 64, 0, 1);
    }
#pragma unroll
    for (int mt = 0; mt < 2; ++mt)
#pragma unroll
      for (int nt = 0; nt < 4; ++nt)
        acc[mt][nt] = wmma_bf16(a[mt], bf[nt], acc[mt][nt]);
  }

#pragma unroll
  for (int nt = 0; nt < 4; ++nt) {
    const int ncol0 = n0 + nt * 16;
    const int t = ncol0 / DIMN;            // 0=Q 1=K 2=V (uniform per tile)
    const int h = (ncol0 % DIMN) / HD;
    const int d = (ncol0 % HD) + r16;
    const float bv = bias[ncol0 + r16];
#pragma unroll
    for (int mt = 0; mt < 2; ++mt) {
      const int mbase = m0 + mt * 16 + half * 8;
      const int b = mbase / SEQ, s0 = mbase % SEQ;
      if (t == 2) {
        // V^T: 8 consecutive s positions per lane -> one 16B store
        v8bf pv;
#pragma unroll
        for (int r = 0; r < 8; ++r) pv[r] = f2bf(acc[mt][nt][r] + bv);
        *(v8bf*)(vtb + ((size_t)(b * NH + h) * HD + d) * SEQ + s0) = pv;
      } else {
        __bf16* dstp = (t == 0 ? qb : kb) +
                       ((size_t)(b * NH + h) * SEQ + s0) * HD + d;
#pragma unroll
        for (int r = 0; r < 8; ++r)
          dstp[(size_t)r * HD] = f2bf(acc[mt][nt][r] + bv);
      }
    }
  }
}

// ---------------------------------------------------------------------------
// Kernel 2: flash attention per (b,h). Block = 8 waves sharing one (b,h).
// K tiles staged by the Tensor Data Mover (wave 0 issues TENSOR_LOAD_TO_LDS,
// TENSORcnt); V^T tiles staged with global_load_async_to_lds_b128 (ASYNCcnt).
// Double-buffered; barriers publish tiles across waves.
// ---------------------------------------------------------------------------
__global__ __launch_bounds__(256) void attn_kernel(
    const __bf16* __restrict__ q, const __bf16* __restrict__ k,
    const __bf16* __restrict__ vt, __bf16* __restrict__ o) {
  __shared__ __align__(16) __bf16 kbuf[2][32 * HD];   // 2 x 4KB  (keys x d)
  __shared__ __align__(16) __bf16 vbuf[2][HD * 32];   // 2 x 4KB  (d x keys)
  __shared__ __align__(16) __bf16 plds[8][16 * 32];   // 8KB      (P staging)

  const int tid = threadIdx.x;
  const int lane = tid & 31, wave = tid >> 5;
  const int r16 = lane & 15, half = lane >> 4;
  const int bh = blockIdx.x;
  const int b = bh / NH, h = bh % NH;
  const int qs0 = (blockIdx.y * 8 + wave) * 16;
  const __bf16* qp = q + (size_t)bh * SEQ * HD;
  const __bf16* kp = k + (size_t)bh * SEQ * HD;
  const __bf16* vp = vt + (size_t)bh * HD * SEQ;
  __bf16* pl = plds[wave];

  // cooperative V^T staging addresses (one 16B async copy per thread)
  const int vrow = tid >> 2, vcol = (tid & 3) * 8;    // 64 rows x 32 cols

  v16bf aq[2];
#pragma unroll
  for (int c = 0; c < 2; ++c)
    aq[c] = fragA_bf16(qp + (size_t)(qs0 + r16) * HD, c * 32, half);

  v8f acc[4] = {};
  float mrow[8], lrow[8];
#pragma unroll
  for (int r = 0; r < 8; ++r) { mrow[r] = -1e30f; lrow[r] = 0.f; }

  // prime buffer 0 with tile kt=0
  if (wave == 0)
    tdm_load_2d((unsigned)(size_t)&kbuf[0][0], kp, HD, SEQ, HD, HD, 32);
  async_cp16(&vbuf[0][vrow * 32 + vcol], vp + (size_t)vrow * SEQ + vcol);

  for (int kt = 0; kt < SEQ; kt += 32) {
    const int cur = (kt >> 5) & 1;
    wait_async0();                                   // own V^T copies done
    if (wave == 0) __builtin_amdgcn_s_wait_tensorcnt(0);  // K TDM done
    __syncthreads();                                 // publish buf[cur]

    if (kt + 32 < SEQ) {  // stream next tile into the other buffer
      const int nxt = cur ^ 1;
      if (wave == 0)
        tdm_load_2d((unsigned)(size_t)&kbuf[nxt][0],
                    kp + (size_t)(kt + 32) * HD, HD, SEQ, HD, HD, 32);
      async_cp16(&vbuf[nxt][vrow * 32 + vcol],
                 vp + (size_t)vrow * SEQ + (kt + 32) + vcol);
    }

    // ---- scores: S[16q x 32k] = Q(16x64) @ K^T  (K from LDS) ----
    v8f s0 = {}, s1 = {};
#pragma unroll
    for (int c = 0; c < 2; ++c) {
      v16bf b0 = fragB_bf16(&kbuf[cur][(r16) * HD], c * 32, half);
      v16bf b1 = fragB_bf16(&kbuf[cur][(16 + r16) * HD], c * 32, half);
      s0 = wmma_bf16(aq[c], b0, s0);
      s1 = wmma_bf16(aq[c], b1, s1);
    }
    float t[8], p0[8], p1[8];
#pragma unroll
    for (int r = 0; r < 8; ++r) {
      s0[r] *= 0.125f;  // 1/sqrt(64)
      s1[r] *= 0.125f;
      t[r] = fmaxf(s0[r], s1[r]);
    }
    // row max across the 16 lanes of each half (rows live per-VGPR)
#pragma unroll
    for (int m = 8; m >= 1; m >>= 1)
#pragma unroll
      for (int r = 0; r < 8; ++r)
        t[r] = fmaxf(t[r], __shfl_xor(t[r], m, 32));
#pragma unroll
    for (int r = 0; r < 8; ++r) {
      const float mn = fmaxf(mrow[r], t[r]);
      const float f = __expf(mrow[r] - mn);
      mrow[r] = mn;
      lrow[r] *= f;
#pragma unroll
      for (int nt = 0; nt < 4; ++nt) acc[nt][r] *= f;
      p0[r] = __expf(s0[r] - mn);
      p1[r] = __expf(s1[r] - mn);
      t[r] = p0[r] + p1[r];
    }
#pragma unroll
    for (int m = 8; m >= 1; m >>= 1)
#pragma unroll
      for (int r = 0; r < 8; ++r)
        t[r] += __shfl_xor(t[r], m, 32);
#pragma unroll
    for (int r = 0; r < 8; ++r) lrow[r] += t[r];

    // ---- stage P (C layout) to LDS, reload as 16x32 A fragment ----
#pragma unroll
    for (int r = 0; r < 8; ++r) {
      pl[(r + half * 8) * 32 + r16] = f2bf(p0[r]);
      pl[(r + half * 8) * 32 + r16 + 16] = f2bf(p1[r]);
    }
    v16bf pa = fragA_bf16(pl + r16 * 32, 0, half);

    // ---- acc += P(16x32) @ V(32x64)  (V^T from LDS) ----
#pragma unroll
    for (int nt = 0; nt < 4; ++nt) {
      v16bf bv = fragB_bf16(&vbuf[cur][(nt * 16 + r16) * 32], 0, half);
      acc[nt] = wmma_bf16(pa, bv, acc[nt]);
    }

    __syncthreads();      // everyone done reading buf[cur] before reuse
  }

  // normalize rows, write bf16 [B, S, D]
#pragma unroll
  for (int r = 0; r < 8; ++r) {
    const float inv = 1.f / lrow[r];
    const int s = qs0 + r + half * 8;
#pragma unroll
    for (int nt = 0; nt < 4; ++nt) {
      const int dd = h * HD + nt * 16 + r16;
      o[((size_t)b * SEQ + s) * DIMN + dd] = f2bf(acc[nt][r] * inv);
    }
  }
}

// ---------------------------------------------------------------------------
// Kernel 3: out = attn(bf16) @ out_w^T(bf16) + out_b   (fp32 output)
// ---------------------------------------------------------------------------
__global__ __launch_bounds__(256) void outproj_kernel(
    const __bf16* __restrict__ a, const __bf16* __restrict__ w,
    const float* __restrict__ bias, float* __restrict__ out) {
  const int lane = threadIdx.x & 31, wave = threadIdx.x >> 5;
  const int r16 = lane & 15, half = lane >> 4;
  const int m0 = blockIdx.y * 256 + wave * 32;
  const int n0 = blockIdx.x * 64;

  const __bf16* arow[2];
  const __bf16* brow[4];
#pragma unroll
  for (int mt = 0; mt < 2; ++mt)
    arow[mt] = a + (size_t)(m0 + mt * 16 + r16) * DIMN;
#pragma unroll
  for (int nt = 0; nt < 4; ++nt)
    brow[nt] = w + (size_t)(n0 + nt * 16 + r16) * DIMN;

  v8f acc[2][4] = {};
  for (int k0 = 0; k0 < DIMN; k0 += 32) {
    v16bf av[2], bf[4];
#pragma unroll
    for (int mt = 0; mt < 2; ++mt) {
      av[mt] = fragA_bf16(arow[mt], k0, half);
      __builtin_prefetch(arow[mt] + k0 + 64, 0, 1);
    }
#pragma unroll
    for (int nt = 0; nt < 4; ++nt) {
      bf[nt] = fragB_bf16(brow[nt], k0, half);
      __builtin_prefetch(brow[nt] + k0 + 64, 0, 1);
    }
#pragma unroll
    for (int mt = 0; mt < 2; ++mt)
#pragma unroll
      for (int nt = 0; nt < 4; ++nt)
        acc[mt][nt] = wmma_bf16(av[mt], bf[nt], acc[mt][nt]);
  }

#pragma unroll
  for (int nt = 0; nt < 4; ++nt) {
    const int n = n0 + nt * 16 + r16;
    const float bv = bias[n];
#pragma unroll
    for (int mt = 0; mt < 2; ++mt) {
#pragma unroll
      for (int r = 0; r < 8; ++r) {
        const int m = m0 + mt * 16 + r + half * 8;
        out[(size_t)m * DIMN + n] = acc[mt][nt][r] + bv;
      }
    }
  }
}

extern "C" void kernel_launch(void* const* d_in, const int* in_sizes, int n_in,
                              void* d_out, int out_size, void* d_ws, size_t ws_size,
                              hipStream_t stream) {
  const float* x     = (const float*)d_in[0];
  const float* qkv_w = (const float*)d_in[1];
  const float* qkv_b = (const float*)d_in[2];
  const float* out_w = (const float*)d_in[3];
  const float* out_b = (const float*)d_in[4];
  float* out = (float*)d_out;

  const size_t nX  = (size_t)BATCH * SEQ * DIMN;   // 8388608
  const size_t nQW = (size_t)3 * DIMN * DIMN;      // 3145728
  const size_t nOW = (size_t)DIMN * DIMN;          // 1048576
  const size_t per = (size_t)BATCH * NH * SEQ * HD;

  __bf16* xb  = (__bf16*)d_ws;
  __bf16* qwb = xb + nX;
  __bf16* owb = qwb + nQW;
  __bf16* qb  = owb + nOW;
  __bf16* kb  = qb + per;
  __bf16* vtb = kb + per;
  __bf16* ao  = vtb + per;   // attn output bf16 [B,S,D]

  // one-shot bf16 conversions (8 elements/thread)
  cvt_kernel<<<dim3(nX / (256 * 8)), 256, 0, stream>>>(x, xb);
  cvt_kernel<<<dim3(nQW / (256 * 8)), 256, 0, stream>>>(qkv_w, qwb);
  cvt_kernel<<<dim3(nOW / (256 * 8)), 256, 0, stream>>>(out_w, owb);

  // qkv: M=8192 N=3072 K=1024; block covers 256x64
  qkv_kernel<<<dim3(3 * DIMN / 64, BATCH * SEQ / 256), 256, 0, stream>>>(
      xb, qwb, qkv_b, qb, kb, vtb);
  // attention: 64 (b,h) pairs x 128 q-tiles (8 per block)
  attn_kernel<<<dim3(BATCH * NH, SEQ / (16 * 8)), 256, 0, stream>>>(
      qb, kb, vtb, ao);
  // out projection: M=8192 N=1024 K=1024
  outproj_kernel<<<dim3(DIMN / 64, BATCH * SEQ / 256), 256, 0, stream>>>(
      ao, owb, out_b, out);
}
